// GeometricAttention_50208167690781
// MI455X (gfx1250) — compile-verified, matrix operand order checked by reading, not run on hardware
//
#include <hip/hip_runtime.h>
#include <hip/hip_bf16.h>

// ---------------------------------------------------------------------------
// GeometricAttention on MI455X (gfx1250, wave32, WMMA bf16 16x16x32)
// Pipeline: cvt -> QKV GEMMs (WMMA) -> flash attention w/ SDF-MLP bias (WMMA)
//           -> output projection GEMM (WMMA)
// Data movement: double-buffered GLOBAL_LOAD_ASYNC_TO_LDS_B128 (ASYNCcnt)
// Workspace requirement: ~26 MB.
// ---------------------------------------------------------------------------

typedef __attribute__((ext_vector_type(16))) __bf16 v16bf;
typedef __attribute__((ext_vector_type(8)))  __bf16 v8bf;
typedef __attribute__((ext_vector_type(8)))  float  v8f;

#define DIM   512
#define NH    8
#define HD    64
#define NQ    2048
#define NKV   2048
#define BATCH 2
#define MROWS (BATCH * NQ)   // 4096
#define SCALE 0.125f         // 64^-0.5

// Build a 16-wide bf16 fragment from two contiguous 8-element (16B) chunks.
// Matches ISA 16-bit A/B layout: lane holds K=(lane/16)*8..+7 and +16..+23.
static __device__ __forceinline__ v16bf make_frag(const __bf16* p0, const __bf16* p1) {
  v8bf lo = *(const v8bf*)p0;
  v8bf hi = *(const v8bf*)p1;
  return __builtin_shufflevector(lo, hi, 0,1,2,3,4,5,6,7,8,9,10,11,12,13,14,15);
}

// Async global->LDS 16B copy (GV mode). LDS offset = low 32 bits of the
// generic __shared__ pointer. Tracked by ASYNCcnt.
static __device__ __forceinline__ void async_copy_b128(const void* gptr, void* lptr) {
  unsigned loff = (unsigned)(unsigned long long)lptr;
  unsigned long long g = (unsigned long long)gptr;
  asm volatile("global_load_async_to_lds_b128 %0, %1, off"
               :: "v"(loff), "v"(g) : "memory");
}
static __device__ __forceinline__ void wait_async0() {
  asm volatile("s_wait_asynccnt 0x0" ::: "memory");
}

// ---------------------------------------------------------------------------
// Conversion kernels
// ---------------------------------------------------------------------------
__global__ __launch_bounds__(256) void cvt_bf16(const float* __restrict__ in,
                                                __bf16* __restrict__ out, int n) {
  int i = blockIdx.x * 256 + threadIdx.x;
  if (i < n) out[i] = (__bf16)in[i];
}

// W (512x512, k-major rows) -> Wt bf16 (n-major rows): Wt[n][k] = W[k][n]
__global__ __launch_bounds__(256) void transpose_cvt512(const float* __restrict__ W,
                                                        __bf16* __restrict__ Wt) {
  int i = blockIdx.x * 256 + threadIdx.x;   // 0..262143, coalesced read
  int k = i >> 9, n = i & 511;
  Wt[(size_t)n * DIM + k] = (__bf16)W[i];
}

// ---------------------------------------------------------------------------
// bf16 GEMM:  C(M x 512) = A(M x 512) @ Wt^T + bias
//   Block = 64x64 tile, 8 waves, each wave -> 16x32 (two 16x16 WMMA accums).
//   K tiles (64x32) ping-pong through LDS via async copies.
// MODE 0: bf16 out, head-split row-major  [b][h][n][dd]
// MODE 1: bf16 out, head-split transposed [b][h][dd][n]   (for V)
// MODE 2: f32 out, plain row-major [m][c]                  (final proj)
// ---------------------------------------------------------------------------
template <int MODE>
static __device__ __forceinline__ void store_out(void* outp, int m, int c, float v) {
  if constexpr (MODE == 2) {
    ((float*)outp)[(size_t)m * DIM + c] = v;
  } else {
    const int b = m >> 11, n = m & 2047;   // Nq = Nk = 2048
    const int h = c >> 6,  dd = c & 63;
    if constexpr (MODE == 0)
      ((__bf16*)outp)[(((size_t)(b * NH + h)) * NQ + n) * HD + dd] = (__bf16)v;
    else
      ((__bf16*)outp)[(((size_t)(b * NH + h)) * HD + dd) * NKV + n] = (__bf16)v;
  }
}

template <int MODE>
__global__ __launch_bounds__(256) void gemm512(const __bf16* __restrict__ A,
                                               const __bf16* __restrict__ Bt,
                                               const float* __restrict__ bias,
                                               void* __restrict__ outp) {
  __shared__ __bf16 Asub[2][64 * 40];   // 64 rows x 32 k, stride 40 (16B aligned)
  __shared__ __bf16 Bsub[2][64 * 40];

  const int tid  = threadIdx.x;
  const int lane = tid & 31;
  const int wave = tid >> 5;
  const int half = lane >> 4;        // lanes 16-31 -> +8 rows / K 8..15,24..31 chunks
  const int idx  = lane & 15;
  const int rt   = wave >> 1;        // row tile 0..3
  const int cp   = wave & 1;         // col pair 0..1
  const int mbase = blockIdx.x * 64;
  const int nbase = blockIdx.y * 64;

  const int lr = tid >> 2;           // cooperative copy: one 16B chunk per thread
  const int lc = (tid & 3) * 8;

  auto issue_tiles = [&](int buf, int kb) {
    async_copy_b128(&A[(size_t)(mbase + lr) * DIM + kb + lc], &Asub[buf][lr * 40 + lc]);
    async_copy_b128(&Bt[(size_t)(nbase + lr) * DIM + kb + lc], &Bsub[buf][lr * 40 + lc]);
  };

  v8f acc0 = {}, acc1 = {};
  issue_tiles(0, 0);

  for (int kb = 0; kb < DIM; kb += 32) {
    const int cur = (kb >> 5) & 1;
    wait_async0();                     // this wave's async copies landed in LDS
    __syncthreads();                   // all waves' copies landed; prev tile free
    if (kb + 32 < DIM) issue_tiles(cur ^ 1, kb + 32);

    const __bf16* ap  = &Asub[cur][(rt * 16 + idx) * 40 + half * 8];
    const __bf16* bp0 = &Bsub[cur][(cp * 32 + idx) * 40 + half * 8];
    const __bf16* bp1 = &Bsub[cur][(cp * 32 + 16 + idx) * 40 + half * 8];
    v16bf af  = make_frag(ap,  ap  + 16);
    v16bf bf0 = make_frag(bp0, bp0 + 16);
    v16bf bf1 = make_frag(bp1, bp1 + 16);
    acc0 = __builtin_amdgcn_wmma_f32_16x16x32_bf16(false, af, false, bf0, (short)0, acc0, false, false);
    acc1 = __builtin_amdgcn_wmma_f32_16x16x32_bf16(false, af, false, bf1, (short)0, acc1, false, false);
  }

  // C layout: VGPR r, lanes 0-15 -> M=r, lanes 16-31 -> M=r+8; N = lane&15
  const int c0 = nbase + cp * 32 + idx;
  const float bb0 = bias[c0], bb1 = bias[c0 + 16];
#pragma unroll
  for (int r = 0; r < 8; ++r) {
    const int m = mbase + rt * 16 + half * 8 + r;
    store_out<MODE>(outp, m, c0,      acc0[r] + bb0);
    store_out<MODE>(outp, m, c0 + 16, acc1[r] + bb1);
  }
}

// ---------------------------------------------------------------------------
// Flash attention with SDF-MLP score bias.
//   Grid: (B*H, NQ/128). Block: 256 thr (8 waves). Wave w: 16 q-rows.
//   K tile (64 kv x 64d) and V^T tile (64d x 64 kv) ping-pong in LDS via
//   async copies; compute on tile t overlaps the copy of tile t+1.
// ---------------------------------------------------------------------------
__global__ __launch_bounds__(256) void flash_geom(
    const __bf16* __restrict__ Q,    // [B*H][NQ][64]
    const __bf16* __restrict__ Kb,   // [B*H][NKV][64]
    const __bf16* __restrict__ Vt,   // [B*H][64][NKV]
    const float* __restrict__ sdf_q, const float* __restrict__ sdf_k,
    const float* __restrict__ W1, const float* __restrict__ b1,
    const float* __restrict__ W2, const float* __restrict__ b2,
    __bf16* __restrict__ attn)       // [B][NQ][512]
{
  __shared__ __bf16 Kl[2][64 * 72];
  __shared__ __bf16 Vl[2][64 * 72];
  __shared__ __bf16 Pl[8][16 * 72];  // per-wave private P staging

  const int tid  = threadIdx.x;
  const int lane = tid & 31, wave = tid >> 5;
  const int half = lane >> 4, idx = lane & 15;
  const int bh = blockIdx.x;               // b*8 + h
  const int b  = bh >> 3, h = bh & 7;
  const int qbase = blockIdx.y * 128 + wave * 16;

  // Tiny MLP params -> SGPRs (uniform across lanes).
  float w1v[16], b1v[16], w2v[16];
#pragma unroll
  for (int u = 0; u < 16; ++u) { w1v[u] = W1[u]; b1v[u] = b1[u]; w2v[u] = W2[u * NH + h]; }
  const float b2v = b2[h];

  // Q as two 16x32 A-fragments (K-dim = d = 64).
  const __bf16* qp = Q + ((size_t)bh * NQ + qbase) * HD;
  v16bf qf[2];
#pragma unroll
  for (int kk = 0; kk < 2; ++kk) {
    const __bf16* p = qp + idx * HD + kk * 32 + half * 8;
    qf[kk] = make_frag(p, p + 16);
  }

  float dq[8];
#pragma unroll
  for (int r = 0; r < 8; ++r) dq[r] = sdf_q[(size_t)b * NQ + qbase + half * 8 + r];

  float mrow[8], lrow[8];
  v8f accO[4] = {{}, {}, {}, {}};
#pragma unroll
  for (int r = 0; r < 8; ++r) { mrow[r] = -1e30f; lrow[r] = 0.f; }

  // 512 x 16B chunks per 64x64 tile -> 2 chunks per thread per tile.
  auto issue_tiles = [&](int buf, int kt) {
#pragma unroll
    for (int e0 = 0; e0 < 2; ++e0) {
      const int e = tid + e0 * 256;
      const int r = e >> 3, c = (e & 7) * 8;
      async_copy_b128(&Kb[((size_t)bh * NKV + kt * 64 + r) * HD + c], &Kl[buf][r * 72 + c]);
      async_copy_b128(&Vt[((size_t)bh * HD + r) * NKV + kt * 64 + c], &Vl[buf][r * 72 + c]);
    }
  };

  issue_tiles(0, 0);

  for (int kt = 0; kt < NKV / 64; ++kt) {
    const int cur = kt & 1;
    wait_async0();
    __syncthreads();
    if (kt + 1 < NKV / 64) issue_tiles(cur ^ 1, kt + 1);

    // S = Q @ K^T * SCALE - mlp(|dq-dk|), four 16x16 col tiles
    v8f S[4];
#pragma unroll
    for (int j = 0; j < 4; ++j) {
      v8f s = {};
#pragma unroll
      for (int kk = 0; kk < 2; ++kk) {
        const __bf16* p = &Kl[cur][(j * 16 + idx) * 72 + kk * 32 + half * 8];
        v16bf kf = make_frag(p, p + 16);
        s = __builtin_amdgcn_wmma_f32_16x16x32_bf16(false, qf[kk], false, kf, (short)0, s, false, false);
      }
      const float dk = sdf_k[(size_t)b * NKV + kt * 64 + j * 16 + idx];
#pragma unroll
      for (int r = 0; r < 8; ++r) {
        const float t = fabsf(dq[r] - dk);
        float f = b2v;
#pragma unroll
        for (int u = 0; u < 16; ++u)
          f += w2v[u] * fmaxf(fmaf(t, w1v[u], b1v[u]), 0.f);
        S[j][r] = s[r] * SCALE - f;
      }
    }

    // Online softmax: rows live on fixed VGPR slots; cols span 16-lane halves.
    float sc[8];
#pragma unroll
    for (int r = 0; r < 8; ++r) {
      float mx = fmaxf(fmaxf(S[0][r], S[1][r]), fmaxf(S[2][r], S[3][r]));
#pragma unroll
      for (int off = 8; off >= 1; off >>= 1) mx = fmaxf(mx, __shfl_xor(mx, off, 32));
      const float mn = fmaxf(mrow[r], mx);
      sc[r] = __expf(mrow[r] - mn);
      mrow[r] = mn;
    }
#pragma unroll
    for (int j = 0; j < 4; ++j)
#pragma unroll
      for (int r = 0; r < 8; ++r) S[j][r] = __expf(S[j][r] - mrow[r]);
#pragma unroll
    for (int r = 0; r < 8; ++r) {
      float sm = S[0][r] + S[1][r] + S[2][r] + S[3][r];
#pragma unroll
      for (int off = 8; off >= 1; off >>= 1) sm += __shfl_xor(sm, off, 32);
      lrow[r] = lrow[r] * sc[r] + sm;
#pragma unroll
      for (int nn = 0; nn < 4; ++nn) accO[nn][r] *= sc[r];
    }

    // P (C-layout) -> LDS row-major bf16 -> reload as A-fragments (wave-private).
    __bf16* pw = &Pl[wave][0];
#pragma unroll
    for (int j = 0; j < 4; ++j)
#pragma unroll
      for (int r = 0; r < 8; ++r)
        pw[(half * 8 + r) * 72 + j * 16 + idx] = (__bf16)S[j][r];
    asm volatile("s_wait_dscnt 0x0" ::: "memory");   // same-wave LDS RAW
    v16bf pf[2];
#pragma unroll
    for (int kk = 0; kk < 2; ++kk) {
      const __bf16* p = &pw[idx * 72 + kk * 32 + half * 8];
      pf[kk] = make_frag(p, p + 16);
    }

    // O += P @ V  (B-fragment cols = d, K-runs contiguous in V^T rows)
#pragma unroll
    for (int nn = 0; nn < 4; ++nn)
#pragma unroll
      for (int kk = 0; kk < 2; ++kk) {
        const __bf16* p = &Vl[cur][(nn * 16 + idx) * 72 + kk * 32 + half * 8];
        v16bf vf = make_frag(p, p + 16);
        accO[nn] = __builtin_amdgcn_wmma_f32_16x16x32_bf16(false, pf[kk], false, vf, (short)0, accO[nn], false, false);
      }
  }

  // Normalize and write bf16 head-interleaved [b][q][h*64+dd] for final GEMM.
#pragma unroll
  for (int r = 0; r < 8; ++r) {
    const float inv = 1.f / lrow[r];
#pragma unroll
    for (int nn = 0; nn < 4; ++nn)
      attn[((size_t)b * NQ + qbase + half * 8 + r) * DIM + h * HD + nn * 16 + idx] =
          (__bf16)(accO[nn][r] * inv);
  }
}

// ---------------------------------------------------------------------------
extern "C" void kernel_launch(void* const* d_in, const int* in_sizes, int n_in,
                              void* d_out, int out_size, void* d_ws, size_t ws_size,
                              hipStream_t stream) {
  (void)in_sizes; (void)n_in; (void)out_size; (void)ws_size;
  const float* x_q  = (const float*)d_in[0];
  const float* x_kv = (const float*)d_in[1];
  const float* sdfq = (const float*)d_in[2];
  const float* sdfk = (const float*)d_in[3];
  const float* Wq = (const float*)d_in[4];  const float* bq = (const float*)d_in[5];
  const float* Wk = (const float*)d_in[6];  const float* bk = (const float*)d_in[7];
  const float* Wv = (const float*)d_in[8];  const float* bv = (const float*)d_in[9];
  const float* Wp = (const float*)d_in[10]; const float* bp = (const float*)d_in[11];
  const float* W1 = (const float*)d_in[12]; const float* b1 = (const float*)d_in[13];
  const float* W2 = (const float*)d_in[14]; const float* b2 = (const float*)d_in[15];

  char* ws = (char*)d_ws;
  size_t off = 0;
  auto alloc = [&](size_t bytes) -> void* {
    void* p = ws + off;
    off += (bytes + 255) & ~(size_t)255;
    return p;
  };
  __bf16* xq_b   = (__bf16*)alloc((size_t)MROWS * DIM * 2);          // 4 MB
  __bf16* xkv_b  = (__bf16*)alloc((size_t)MROWS * DIM * 2);          // 4 MB
  __bf16* Wqt    = (__bf16*)alloc((size_t)DIM * DIM * 2);            // 0.5 MB
  __bf16* Wkt    = (__bf16*)alloc((size_t)DIM * DIM * 2);
  __bf16* Wvt    = (__bf16*)alloc((size_t)DIM * DIM * 2);
  __bf16* Wpt    = (__bf16*)alloc((size_t)DIM * DIM * 2);
  __bf16* q_ws   = (__bf16*)alloc((size_t)BATCH * NH * NQ  * HD * 2); // 4 MB
  __bf16* k_ws   = (__bf16*)alloc((size_t)BATCH * NH * NKV * HD * 2); // 4 MB
  __bf16* vt_ws  = (__bf16*)alloc((size_t)BATCH * NH * HD * NKV * 2); // 4 MB
  __bf16* attn_b = (__bf16*)alloc((size_t)MROWS * DIM * 2);           // 4 MB

  const int NE = MROWS * DIM;  // 2,097,152
  cvt_bf16<<<(NE + 255) / 256, 256, 0, stream>>>(x_q,  xq_b,  NE);
  cvt_bf16<<<(NE + 255) / 256, 256, 0, stream>>>(x_kv, xkv_b, NE);
  transpose_cvt512<<<DIM * DIM / 256, 256, 0, stream>>>(Wq, Wqt);
  transpose_cvt512<<<DIM * DIM / 256, 256, 0, stream>>>(Wk, Wkt);
  transpose_cvt512<<<DIM * DIM / 256, 256, 0, stream>>>(Wv, Wvt);
  transpose_cvt512<<<DIM * DIM / 256, 256, 0, stream>>>(Wp, Wpt);

  dim3 gg(MROWS / 64, DIM / 64);   // 64 x 8 blocks
  gemm512<0><<<gg, 256, 0, stream>>>(xq_b,  Wqt, bq, q_ws);
  gemm512<0><<<gg, 256, 0, stream>>>(xkv_b, Wkt, bk, k_ws);
  gemm512<1><<<gg, 256, 0, stream>>>(xkv_b, Wvt, bv, vt_ws);

  flash_geom<<<dim3(BATCH * NH, NQ / 128), 256, 0, stream>>>(
      q_ws, k_ws, vt_ws, sdfq, sdfk, W1, b1, W2, b2, attn_b);

  gemm512<2><<<gg, 256, 0, stream>>>(attn_b, Wpt, bp, d_out);
}
